// Canary_Model_12051678232790
// MI455X (gfx1250) — matrix operand order repaired and checked by reading, not compiled, not run on hardware
//
#include <hip/hip_runtime.h>
#include <stdint.h>

typedef __attribute__((ext_vector_type(16))) _Float16 v16h;
typedef __attribute__((ext_vector_type(8)))  float    v8f;

// ---------------- constants ----------------
#define NB    256      // batch
#define CIN   200
#define PP    13
#define PIXN  169
#define C8N   1600     // 8*CIN
#define FD    128
#define SLEN  28       // scan length
// mamba1 (dm=128): d_inner=1024 nh=64 conv_dim=1056 nproj=2144 dtoff=2080
// mamba2 (dm=28):  d_inner=224  nh=14 conv_dim=256  nproj=494  dtoff=480, L=128
// mamba3 (dm=200): d_inner=1600 nh=100 conv_dim=1632 nproj=3332 dtoff=3232, L=1

static __device__ __forceinline__ float siluf(float x){ return x / (1.0f + __expf(-x)); }

// integer value of an AS(3) pointer == workgroup-relative LDS byte offset
static __device__ __forceinline__ uint32_t lds_addr_of(void* p) {
    return (uint32_t)(uintptr_t)(__attribute__((address_space(3))) void*)p;
}

// ---------------- per-sample normalize ----------------
__global__ void k_norm(const float* __restrict__ x, float* __restrict__ xn)
{
    const int b = blockIdx.x, t = threadIdx.x;
    const int N = CIN * PIXN;
    const float* xb = x + (size_t)b * N;
    float s = 0.f, s2 = 0.f;
    for (int i = t; i < N; i += 256) { float v = xb[i]; s += v; s2 += v * v; }
    __shared__ float rs[256], rq[256];
    __shared__ float mu_s, inv_s;
    rs[t] = s; rq[t] = s2; __syncthreads();
    for (int k = 128; k > 0; k >>= 1) { if (t < k) { rs[t] += rs[t+k]; rq[t] += rq[t+k]; } __syncthreads(); }
    if (t == 0) {
        float mu = rs[0] / N;
        float var = fmaxf(rq[0] / N - mu * mu, 0.f);
        mu_s = mu; inv_s = 1.f / (sqrtf(var) + 1e-8f);
    }
    __syncthreads();
    float mu = mu_s, inv = inv_s;
    float* on = xn + (size_t)b * N;
    for (int i = t; i < N; i += 256) on[i] = (xb[i] - mu) * inv;
}

// ---------------- 7x3x3 conv + bn3a + silu ----------------
__global__ void k_c3a(const float* __restrict__ xn, const float* __restrict__ w,
                      const float* __restrict__ cb, const float* __restrict__ bn,
                      float* __restrict__ w1, size_t total)
{
    size_t idx = (size_t)blockIdx.x * 256 + threadIdx.x;
    if (idx >= total) return;
    int pix = idx % PIXN; size_t r = idx / PIXN;
    int d = r % CIN; int b = r / CIN;
    int y = pix / PP, x = pix % PP;
    const float* xb = xn + (size_t)b * CIN * PIXN;
    float acc = cb[0];
    for (int kd = 0; kd < 7; ++kd) {
        int dd = d + kd - 3; if (dd < 0 || dd >= CIN) continue;
        for (int ky = 0; ky < 3; ++ky) {
            int yy = y + ky - 1; if (yy < 0 || yy >= PP) continue;
            for (int kx = 0; kx < 3; ++kx) {
                int xx = x + kx - 1; if (xx < 0 || xx >= PP) continue;
                acc += w[(kd*3+ky)*3+kx] * xb[(size_t)dd*PIXN + yy*PP + xx];
            }
        }
    }
    float v = (acc - bn[2]) * rsqrtf(bn[3] + 1e-5f) * bn[0] + bn[1];
    w1[idx] = siluf(v);
}

// ---------------- 1->8 pointwise + bn3b + silu, f16 pixel-major ----------------
__global__ void k_v(const float* __restrict__ w1, const float* __restrict__ cw,
                    const float* __restrict__ cb, const float* __restrict__ bn,
                    _Float16* __restrict__ v, size_t total)
{
    size_t idx = (size_t)blockIdx.x * 256 + threadIdx.x;
    if (idx >= total) return;
    int c = idx % C8N; size_t r = idx / C8N;
    int pix = r % PIXN; int b = r / PIXN;
    int c8 = c / CIN, d = c % CIN;
    float wv = w1[((size_t)b * CIN + d) * PIXN + pix];
    float a = wv * cw[c8] + cb[c8];
    float o = (a - bn[16+c8]) * rsqrtf(bn[24+c8] + 1e-5f) * bn[c8] + bn[8+c8];
    v[idx] = (_Float16)siluf(o);
}

// ---------------- depthwise 3x3 + bn2a + relu -> f16 GEMM-A ----------------
__global__ void k_dw(const _Float16* __restrict__ v, const float* __restrict__ w,
                     const float* __restrict__ cb, const float* __restrict__ bn,
                     _Float16* __restrict__ adw, size_t total)
{
    size_t idx = (size_t)blockIdx.x * 256 + threadIdx.x;
    if (idx >= total) return;
    int c = idx % C8N; size_t r = idx / C8N;
    int pix = r % PIXN; int b = r / PIXN;
    int y = pix / PP, x = pix % PP;
    const _Float16* vb = v + (size_t)b * PIXN * C8N;
    float acc = cb[c];
    for (int ky = 0; ky < 3; ++ky) {
        int yy = y + ky - 1; if (yy < 0 || yy >= PP) continue;
        for (int kx = 0; kx < 3; ++kx) {
            int xx = x + kx - 1; if (xx < 0 || xx >= PP) continue;
            acc += w[c*9 + ky*3 + kx] * (float)vb[(size_t)(yy*PP+xx)*C8N + c];
        }
    }
    float o = (acc - bn[2*C8N+c]) * rsqrtf(bn[3*C8N+c] + 1e-5f) * bn[c] + bn[C8N+c];
    adw[idx] = (_Float16)fmaxf(o, 0.f);
}

// ---------------- pack weights (K,N) into WMMA B-operand layout ----------------
// element (k,n) read from src[k*sk + n*sn]; word = ((ntile*ksteps + ks)*32 + lane)*8 + j
__global__ void k_pack(const float* __restrict__ src, uint32_t* __restrict__ dst,
                       int K, int sk, int sn, size_t total)
{
    size_t widx = (size_t)blockIdx.x * 256 + threadIdx.x;
    if (widx >= total) return;
    int j = widx & 7;
    int lane = (widx >> 3) & 31;
    size_t r = widx >> 8;
    int ksteps = K >> 5;
    int ks = r % ksteps;
    int ntile = r / ksteps;
    int col = ntile * 16 + (lane & 15);
    int k0 = ks * 32 + (lane >> 4) * 16 + 2 * j;
    union { _Float16 h[2]; uint32_t u; } pk;
    pk.h[0] = (_Float16)src[(size_t)k0 * sk + (size_t)col * sn];
    pk.h[1] = (_Float16)src[(size_t)(k0+1) * sk + (size_t)col * sn];
    dst[widx] = pk.u;
}

// ---------------- WMMA f16 GEMM: C(MxN,f32) = A(MxK,f16) * Bpacked ----------------
// block = 8 waves, each wave: one 16-row M-tile x two 16-col N-tiles; grid.y = N/32.
// B panel staged into LDS once per block in 32KB segments via async global->LDS DMA
// (global_load_async_to_lds_b128 + s_wait_asynccnt), then read with ds_load_b128.
// mode 0: raw store; mode 1: relu((acc + bias[n]) * bn_scale + bn_shift)
#define KSEG 16
__global__ void k_wgemm(const _Float16* __restrict__ A, const uint32_t* __restrict__ Bp,
                        float* __restrict__ C, int M, int K, int N,
                        int mode, const float* __restrict__ bias, const float* __restrict__ bn)
{
    __shared__ uint32_t bsh[2 * KSEG * 256];     // 32 KB
    const int lane = threadIdx.x & 31;
    const int wave = threadIdx.x >> 5;
    int mtile = blockIdx.x * 8 + wave;
    const bool valid = (mtile * 16 < M);
    if (!valid) mtile = 0;
    const int nt0 = blockIdx.y * 2;
    const int kh = lane >> 4;
    const _Float16* arow = A + (size_t)(mtile*16 + (lane & 15)) * K + kh * 16;
    const int ksteps = K >> 5;
    v8f acc0 = {}; v8f acc1 = {};

    for (int ks0 = 0; ks0 < ksteps; ks0 += KSEG) {
        const int kseg = (ksteps - ks0 < KSEG) ? (ksteps - ks0) : KSEG;
        __syncthreads();   // previous segment fully consumed
        // async-stage both 16-col B panels for this K segment (contiguous in packed layout)
        #pragma unroll
        for (int nt = 0; nt < 2; ++nt) {
            const uint32_t* g = Bp + ((size_t)(nt0 + nt) * ksteps + ks0) * 256;
            uint32_t* l = bsh + nt * KSEG * 256;
            for (int i = threadIdx.x; i < kseg * 64; i += 256) {   // 16B chunks
                uint32_t la = lds_addr_of(l + (size_t)i * 4);
                unsigned long long ga = (unsigned long long)(uintptr_t)(g + (size_t)i * 4);
                asm volatile("global_load_async_to_lds_b128 %0, %1, off"
                             :: "v"(la), "v"(ga) : "memory");
            }
        }
        asm volatile("s_wait_asynccnt 0" ::: "memory");
        __syncthreads();   // all waves' DMA chunks visible

        for (int kk = 0; kk < kseg; ++kk) {
            const int ks = ks0 + kk;
            union { uint4 q[2]; v16h v; } ua, ub0, ub1;
            ua.q[0] = *(const uint4*)(arow + ks*32);
            ua.q[1] = *(const uint4*)(arow + ks*32 + 8);
            const uint32_t* p0 = bsh + (size_t)kk*256 + lane*8;
            const uint32_t* p1 = bsh + (size_t)(KSEG + kk)*256 + lane*8;
            ub0.q[0] = *(const uint4*)(p0);     ub0.q[1] = *(const uint4*)(p0 + 4);
            ub1.q[0] = *(const uint4*)(p1);     ub1.q[1] = *(const uint4*)(p1 + 4);
            if (ks + 1 < ksteps)
                __builtin_prefetch(arow + (ks+1)*32, 0, 3);   // global_prefetch_b8
            acc0 = __builtin_amdgcn_wmma_f32_16x16x32_f16(false, ua.v, false, ub0.v,
                                                          (short)0, acc0, false, false);
            acc1 = __builtin_amdgcn_wmma_f32_16x16x32_f16(false, ua.v, false, ub1.v,
                                                          (short)0, acc1, false, false);
        }
    }

    if (!valid) return;
    const int rbase = mtile * 16 + kh * 8;
    #pragma unroll
    for (int nt = 0; nt < 2; ++nt) {
        v8f a = nt ? acc1 : acc0;
        int n = (nt0 + nt) * 16 + (lane & 15);
        float sc = 1.f, sh = 0.f, bi = 0.f;
        if (mode == 1) {
            bi = bias[n];
            float g = bn[n], be = bn[N+n], mm = bn[2*N+n], va = bn[3*N+n];
            sc = g * rsqrtf(va + 1e-5f);
            sh = be - mm * sc;
        }
        #pragma unroll
        for (int r = 0; r < 8; ++r) {
            float val = a[r];
            if (mode == 1) val = fmaxf((val + bi) * sc + sh, 0.f);
            C[(size_t)(rbase + r) * N + n] = val;
        }
    }
}

// ---------------- small_scan: f(B,169,128) -> s(B,28,128) f16 ----------------
__global__ void k_sscan(const float* __restrict__ fpm, _Float16* __restrict__ s16)
{
    int b = blockIdx.x, t = blockIdx.y, ch = threadIdx.x;
    const float* f = fpm + (size_t)b * PIXN * FD;
    const int c = PP / 2;
    float val = 0.f;
    if (t == 0) val = f[(c*PP+c)*FD + ch];
    else if (t >= 4) {
        int i = (t - 4) >> 2, side = (t - 4) & 3;
        int n = 2*i + 3;
        float ctr = 0.5f * (n - 1), ws = 0.f, acc = 0.f;
        for (int j = 0; j < n; ++j) {
            float w = 1.f / (1.f + fabsf((float)j - ctr));
            ws += w;
            int yy, xx;
            if      (side == 0) { yy = c-i-1;   xx = c-i-1+j; }
            else if (side == 1) { yy = c-i-1+j; xx = c+i+1;   }
            else if (side == 2) { yy = c+i+1;   xx = c-i-1+j; }
            else                { yy = c-i-1+j; xx = c-i-1;   }
            acc += w * f[(yy*PP+xx)*FD + ch];
        }
        val = acc / ws;
    }
    s16[((size_t)b * SLEN + t) * FD + ch] = (_Float16)val;
}

// ---------------- depthwise causal conv1d (DCONV=4) + silu ----------------
__global__ void k_conv1d(const float* __restrict__ zx, int ld, int off,
                         const float* __restrict__ cw, const float* __restrict__ cb,
                         float* __restrict__ out, int L, int cd, size_t total)
{
    size_t idx = (size_t)blockIdx.x * 256 + threadIdx.x;
    if (idx >= total) return;
    int c = idx % cd; size_t r = idx / cd;
    int l = r % L; int b = r / L;
    float acc = cb[c];
    #pragma unroll
    for (int j = 0; j < 4; ++j) {
        int ll = l - 3 + j;
        if (ll >= 0) acc += cw[c*4 + j] * zx[(size_t)(b*L + ll) * ld + off + c];
    }
    out[idx] = siluf(acc);
}

// ---------------- dt = softplus(dt_raw + dt_bias) ----------------
__global__ void k_dtsp(const float* __restrict__ zx, int ld, int off,
                       const float* __restrict__ dtb, float* __restrict__ out,
                       int nh, size_t total)
{
    size_t idx = (size_t)blockIdx.x * 256 + threadIdx.x;
    if (idx >= total) return;
    int h = idx % nh; size_t row = idx / nh;
    float v = zx[row * ld + off + h] + dtb[h];
    out[idx] = v > 20.f ? v : log1pf(__expf(v));
}

// ---------------- selective state-space scan ----------------
// block(16 p,16 heads); grid(B, ceil(nh/16)); B/C staged in LDS each step
__global__ void k_ssm(const float* __restrict__ xbcs, const float* __restrict__ dtsp,
                      const float* __restrict__ Alog, const float* __restrict__ Dp,
                      float* __restrict__ y, int L, int nh, int cd, int di)
{
    int b = blockIdx.x;
    int p = threadIdx.x, hs = threadIdx.y;
    int h = blockIdx.y * 16 + hs;
    bool act = (h < nh);
    __shared__ float sB[16], sC[16];
    float st[16];
    #pragma unroll
    for (int n = 0; n < 16; ++n) st[n] = 0.f;
    float Ah = act ? -__expf(Alog[h]) : 0.f;
    float Dh = act ? Dp[h] : 0.f;
    int flat = hs * 16 + p;
    for (int l = 0; l < L; ++l) {
        const float* rowp = xbcs + (size_t)(b*L + l) * cd;
        __syncthreads();
        if (flat < 16) sB[flat] = rowp[di + flat];
        else if (flat < 32) sC[flat-16] = rowp[di + 16 + flat - 16];
        __syncthreads();
        if (act) {
            float dtv = dtsp[(size_t)(b*L + l) * nh + h];
            float dA = __expf(dtv * Ah);
            float xv = rowp[h*16 + p];
            float dx = dtv * xv;
            float yv = 0.f;
            #pragma unroll
            for (int n = 0; n < 16; ++n) {
                st[n] = st[n] * dA + dx * sB[n];
                yv += st[n] * sC[n];
            }
            y[(size_t)(b*L + l) * di + h*16 + p] = yv + Dh * xv;
        }
    }
}

// ---------------- gated RMSNorm: y*silu(z), rms, *norm_w -> f32 and/or f16 ----------------
__global__ void k_gate(float* __restrict__ yraw, const float* __restrict__ zx, int ld,
                       const float* __restrict__ nw, float* __restrict__ o32,
                       _Float16* __restrict__ o16, int di)
{
    int row = blockIdx.x, t = threadIdx.x;
    __shared__ float red[256];
    __shared__ float s_scale;
    float* yr = yraw + (size_t)row * di;
    const float* zr = zx + (size_t)row * ld;
    float ss = 0.f;
    for (int i = t; i < di; i += 256) {
        float z = zr[i];
        float v = yr[i] * (z / (1.f + __expf(-z)));
        yr[i] = v;
        ss += v * v;
    }
    red[t] = ss; __syncthreads();
    for (int s = 128; s > 0; s >>= 1) { if (t < s) red[t] += red[t+s]; __syncthreads(); }
    if (t == 0) s_scale = rsqrtf(red[0] / di + 1e-5f);
    __syncthreads();
    float sc = s_scale;
    for (int i = t; i < di; i += 256) {
        float v = yr[i] * sc * nw[i];
        if (o32) o32[(size_t)row * di + i] = v;
        if (o16) o16[(size_t)row * di + i] = (_Float16)v;
    }
}

// ---------------- plain f32 GEMM (+ optional bias) ----------------
__global__ void k_pgemm(const float* __restrict__ A, const float* __restrict__ W,
                        const float* __restrict__ bias, float* __restrict__ C,
                        int K, int N, size_t total)
{
    size_t idx = (size_t)blockIdx.x * 256 + threadIdx.x;
    if (idx >= total) return;
    int n = idx % N; size_t m = idx / N;
    float acc = bias ? bias[n] : 0.f;
    const float* ar = A + m * (size_t)K;
    for (int k = 0; k < K; ++k) acc += ar[k] * W[(size_t)k * N + n];
    C[idx] = acc;
}

// ---------------- y1 (B,28,128) -> x2 (B*128, 28) ----------------
__global__ void k_transpose(const float* __restrict__ y1, float* __restrict__ x2, size_t total)
{
    size_t idx = (size_t)blockIdx.x * 256 + threadIdx.x;
    if (idx >= total) return;
    int l = idx % SLEN; size_t r = idx / SLEN;
    int ch = r % FD; int b = r / FD;
    x2[idx] = y1[((size_t)b * SLEN + l) * FD + ch];
}

// ---------------- x1 = mean over first 4 seq positions of m2 output ----------------
__global__ void k_x1(const float* __restrict__ y2, float* __restrict__ x1, size_t total)
{
    size_t idx = (size_t)blockIdx.x * 256 + threadIdx.x;
    if (idx >= total) return;
    const float* r = y2 + idx * SLEN;
    x1[idx] = 0.25f * (r[0] + r[1] + r[2] + r[3]);
}

// ---------------- extract normalized center pixel ----------------
__global__ void k_cp(const float* __restrict__ xn, float* __restrict__ cp, size_t total)
{
    size_t idx = (size_t)blockIdx.x * 256 + threadIdx.x;
    if (idx >= total) return;
    cp[idx] = xn[idx * PIXN + (PP/2)*PP + PP/2];
}

// ---------------- fused head: gate softmax, MLP, bn ----------------
__global__ void k_head(const float* __restrict__ x1f, const float* __restrict__ x2f,
                       const float* __restrict__ fgw, const float* __restrict__ fgb,
                       const float* __restrict__ c1w, const float* __restrict__ c1b,
                       const float* __restrict__ c2w, const float* __restrict__ c2b,
                       const float* __restrict__ bnc, float* __restrict__ out)
{
    int b = blockIdx.x, t = threadIdx.x; // 128 threads
    __shared__ float xf[128], hb[64], r0[128], r1[128];
    __shared__ float w0s, w1s;
    float a = x1f[b*FD + t], c = x2f[b*FD + t];
    r0[t] = a * fgw[t*2+0] + c * fgw[(FD+t)*2+0];
    r1[t] = a * fgw[t*2+1] + c * fgw[(FD+t)*2+1];
    __syncthreads();
    for (int s = 64; s > 0; s >>= 1) { if (t < s) { r0[t]+=r0[t+s]; r1[t]+=r1[t+s]; } __syncthreads(); }
    if (t == 0) {
        float l0 = r0[0] + fgb[0], l1 = r1[0] + fgb[1];
        float m = fmaxf(l0, l1);
        float e0 = __expf(l0 - m), e1 = __expf(l1 - m);
        w0s = e0 / (e0 + e1); w1s = e1 / (e0 + e1);
    }
    __syncthreads();
    xf[t] = w0s * a + w1s * c;
    __syncthreads();
    if (t < 64) {
        float s = c1b[t];
        for (int i = 0; i < 128; ++i) s += xf[i] * c1w[i*64 + t];
        hb[t] = fmaxf(s, 0.f);
    }
    __syncthreads();
    if (t < 16) {
        float s = c2b[t];
        for (int i = 0; i < 64; ++i) s += hb[i] * c2w[i*16 + t];
        out[b*16 + t] = (s - bnc[32+t]) * rsqrtf(bnc[48+t] + 1e-5f) * bnc[t] + bnc[16+t];
    }
}

// =================================================================
static inline int cdiv(size_t a, int b) { return (int)((a + b - 1) / b); }

extern "C" void kernel_launch(void* const* d_in, const int* in_sizes, int n_in,
                              void* d_out, int out_size, void* d_ws, size_t ws_size,
                              hipStream_t stream)
{
    const float* X     = (const float*)d_in[0];
    const float* C3AW  = (const float*)d_in[1];
    const float* C3AB  = (const float*)d_in[2];
    const float* BN3A  = (const float*)d_in[3];
    const float* C3BW  = (const float*)d_in[4];
    const float* C3BB  = (const float*)d_in[5];
    const float* BN3B  = (const float*)d_in[6];
    const float* C2DW  = (const float*)d_in[7];
    const float* C2DB  = (const float*)d_in[8];
    const float* BN2A  = (const float*)d_in[9];
    const float* C2PW  = (const float*)d_in[10];
    const float* C2PB  = (const float*)d_in[11];
    const float* BN2B  = (const float*)d_in[12];
    const float* LINW  = (const float*)d_in[13];
    const float* LINB  = (const float*)d_in[14];
    const float* FGW   = (const float*)d_in[15];
    const float* FGB   = (const float*)d_in[16];
    const float* CL1W  = (const float*)d_in[17];
    const float* CL1B  = (const float*)d_in[18];
    const float* CL2W  = (const float*)d_in[19];
    const float* CL2B  = (const float*)d_in[20];
    const float* BNC   = (const float*)d_in[21];
    // m1 (22..29), m2 (30..37), m3 (38..45): in_w, conv_w, conv_b, dt_bias, A_log, D, norm_w, out_w
    const float* M1[8]; const float* M2[8]; const float* M3[8];
    for (int i = 0; i < 8; ++i) { M1[i] = (const float*)d_in[22+i];
                                  M2[i] = (const float*)d_in[30+i];
                                  M3[i] = (const float*)d_in[38+i]; }

    // ---- workspace bump allocator + two reused big regions ----
    char* base = (char*)d_ws;
    size_t off = 0;
    auto alloc = [&](size_t n) { size_t o = (off + 255) & ~(size_t)255; off = o + n; return base + o; };
    float*    XN    = (float*)alloc((size_t)NB*CIN*PIXN*4);     // persistent
    char*     R1    = alloc(140ull << 20);
    char*     R2    = alloc(140ull << 20);
    uint32_t* PKC2P = (uint32_t*)alloc(102400*4);
    uint32_t* PKM1I = (uint32_t*)alloc(137216*4);
    uint32_t* PKM1O = (uint32_t*)alloc(65536*4);

    // phase A
    _Float16* VF16 = (_Float16*)R1;                 // (B,169,1600) f16
    float*    W1   = (float*)R2;                    // (B,200,169) f32
    _Float16* ADW  = (_Float16*)R2;                 // aliases W1 (dead by then)
    // phase B (R1 after VF16 dead, R2 after ADW dead)
    float*    FPM   = (float*)R1;                   // (B,169,128)
    _Float16* S16   = (_Float16*)(R1 + 22200000);   // (B*28,128)
    float*    ZX1   = (float*)(R1 + 24100000);      // (7168,2144)
    float*    XBCS1 = (float*)(R1 + 85600000);      // (7168,1056)
    float*    DT1   = (float*)(R1 + 115900000);     // (7168,64)
    float*    YRAW1 = (float*)R2;                   // (7168,1024)
    _Float16* YN1   = (_Float16*)(R2 + 29400000);   // (7168,1024) f16
    float*    Y1    = (float*)(R2 + 44100000);      // (7168,128)
    float*    X2    = (float*)(R2 + 47800000);      // (32768,28)
    float*    ZX2   = (float*)(R2 + 51500000);      // (32768,494)
    // phase C (R1 after m1 done)
    float*    XBCS2 = (float*)R1;                   // (32768,256)
    float*    DT2   = (float*)(R1 + 33600000);
    float*    YRAW2 = (float*)(R1 + 35500000);      // (32768,224)
    float*    YN2   = (float*)(R1 + 64900000);      // (32768,224)
    float*    Y2    = (float*)(R1 + 94300000);      // (32768,28)
    float*    X1F   = (float*)(R1 + 98000000);      // (256,128)
    float*    CP    = (float*)(R1 + 98200000);      // (256,200)
    float*    ZX3   = (float*)(R1 + 98500000);      // (256,3332)
    float*    XBCS3 = (float*)(R1 + 102000000);     // (256,1632)
    float*    DT3   = (float*)(R1 + 103700000);     // (256,100)
    float*    YRAW3 = (float*)(R1 + 103900000);     // (256,1600)
    float*    YN3   = (float*)(R1 + 105600000);     // (256,1600)
    float*    CPO   = (float*)(R1 + 107300000);     // (256,200)
    float*    X2F   = (float*)(R1 + 107600000);     // (256,128)
    float*    OUT   = (float*)d_out;

    // ---- feature extraction ----
    k_norm<<<NB, 256, 0, stream>>>(X, XN);
    { size_t T = (size_t)NB*CIN*PIXN;
      k_c3a<<<cdiv(T,256), 256, 0, stream>>>(XN, C3AW, C3AB, BN3A, W1, T); }
    { size_t T = (size_t)NB*PIXN*C8N;
      k_v <<<cdiv(T,256), 256, 0, stream>>>(W1, C3BW, C3BB, BN3B, VF16, T);
      k_dw<<<cdiv(T,256), 256, 0, stream>>>(VF16, C2DW, C2DB, BN2A, ADW, T); }
    // pack weights: c2p_w is (N=128,K=1600) -> (k,n) at [n*1600+k]
    k_pack<<<cdiv(102400,256), 256, 0, stream>>>(C2PW, PKC2P, 1600, 1, 1600, 102400);
    k_pack<<<cdiv(137216,256), 256, 0, stream>>>(M1[0], PKM1I, 128, 2144, 1, 137216);  // (128,2144)
    k_pack<<<cdiv(65536,256),  256, 0, stream>>>(M1[7], PKM1O, 1024, 128, 1, 65536);   // (1024,128)
    // big GEMM: (43264,1600)x(1600,128), fused bias + bn2b + relu
    k_wgemm<<<dim3(43264/16/8, 128/32), 256, 0, stream>>>(ADW, PKC2P, FPM,
            43264, 1600, 128, 1, C2PB, BN2B);
    // ring scan
    k_sscan<<<dim3(NB, SLEN), FD, 0, stream>>>(FPM, S16);

    // ---- mamba1 (B=256, L=28, dm=128) ----
    k_wgemm<<<dim3(7168/16/8, 2144/32), 256, 0, stream>>>(S16, PKM1I, ZX1,
            7168, 128, 2144, 0, nullptr, nullptr);
    { size_t T = 7168ull*1056;
      k_conv1d<<<cdiv(T,256), 256, 0, stream>>>(ZX1, 2144, 1024, M1[1], M1[2], XBCS1, SLEN, 1056, T); }
    { size_t T = 7168ull*64;
      k_dtsp<<<cdiv(T,256), 256, 0, stream>>>(ZX1, 2144, 2080, M1[3], DT1, 64, T); }
    k_ssm<<<dim3(NB, 4), dim3(16,16), 0, stream>>>(XBCS1, DT1, M1[4], M1[5], YRAW1, SLEN, 64, 1056, 1024);
    k_gate<<<7168, 256, 0, stream>>>(YRAW1, ZX1, 2144, M1[6], nullptr, YN1, 1024);
    k_wgemm<<<dim3(7168/16/8, 128/32), 256, 0, stream>>>(YN1, PKM1O, Y1,
            7168, 1024, 128, 0, nullptr, nullptr);

    // ---- mamba2 (B=256, L=128, dm=28) ----
    { size_t T = 32768ull*28;
      k_transpose<<<cdiv(T,256), 256, 0, stream>>>(Y1, X2, T); }
    { size_t T = 32768ull*494;
      k_pgemm<<<cdiv(T,256), 256, 0, stream>>>(X2, M2[0], nullptr, ZX2, 28, 494, T); }
    { size_t T = 32768ull*256;
      k_conv1d<<<cdiv(T,256), 256, 0, stream>>>(ZX2, 494, 224, M2[1], M2[2], XBCS2, 128, 256, T); }
    { size_t T = 32768ull*14;
      k_dtsp<<<cdiv(T,256), 256, 0, stream>>>(ZX2, 494, 480, M2[3], DT2, 14, T); }
    k_ssm<<<dim3(NB, 1), dim3(16,16), 0, stream>>>(XBCS2, DT2, M2[4], M2[5], YRAW2, 128, 14, 256, 224);
    k_gate<<<32768, 256, 0, stream>>>(YRAW2, ZX2, 494, M2[6], YN2, nullptr, 224);
    { size_t T = 32768ull*28;
      k_pgemm<<<cdiv(T,256), 256, 0, stream>>>(YN2, M2[7], nullptr, Y2, 224, 28, T); }
    k_x1<<<cdiv(32768,256), 256, 0, stream>>>(Y2, X1F, 32768);

    // ---- mamba3 (B=256, L=1, dm=200) on normalized center pixel ----
    k_cp<<<cdiv((size_t)NB*CIN,256), 256, 0, stream>>>(XN, CP, (size_t)NB*CIN);
    { size_t T = 256ull*3332;
      k_pgemm<<<cdiv(T,256), 256, 0, stream>>>(CP, M3[0], nullptr, ZX3, 200, 3332, T); }
    { size_t T = 256ull*1632;
      k_conv1d<<<cdiv(T,256), 256, 0, stream>>>(ZX3, 3332, 1600, M3[1], M3[2], XBCS3, 1, 1632, T); }
    { size_t T = 256ull*100;
      k_dtsp<<<cdiv(T,256), 256, 0, stream>>>(ZX3, 3332, 3232, M3[3], DT3, 100, T); }
    k_ssm<<<dim3(NB, 7), dim3(16,16), 0, stream>>>(XBCS3, DT3, M3[4], M3[5], YRAW3, 1, 100, 1632, 1600);
    k_gate<<<256, 256, 0, stream>>>(YRAW3, ZX3, 3332, M3[6], YN3, nullptr, 1600);
    { size_t T = 256ull*200;
      k_pgemm<<<cdiv(T,256), 256, 0, stream>>>(YN3, M3[7], nullptr, CPO, 1600, 200, T); }
    { size_t T = 256ull*128;
      k_pgemm<<<cdiv(T,256), 256, 0, stream>>>(CPO, LINW, LINB, X2F, 200, 128, T); }

    // ---- head ----
    k_head<<<NB, 128, 0, stream>>>(X1F, X2F, FGW, FGB, CL1W, CL1B, CL2W, CL2B, BNC, OUT);

    (void)in_sizes; (void)n_in; (void)out_size; (void)ws_size;
}
